// DbrxExpertGLU_36627481100905
// MI455X (gfx1250) — compile-verified, matrix operand order checked by reading, not run on hardware
//
#include <hip/hip_runtime.h>

// ---------------------------------------------------------------------------
// SwiGLU expert FFN for MI455X (gfx1250, wave32, WMMA + TDM).
//   pass 1: h = silu(x @ w1) * (x @ v1)   -> bf16 workspace [T, F]
//   pass 2: out = h @ w2^T                -> f32 [T, H]
// bf16 WMMA (16x16x32, f32 accum) = 8x the per-instr FLOPs of f32 WMMA.
// Double-buffered LDS ping-pong (1 barrier / K-step); pass-2 A tiles are
// moved by the Tensor Data Mover (tensor_load_to_lds + s_wait_tensorcnt)
// when the builtins exist, register-relay copy otherwise.
// ---------------------------------------------------------------------------

typedef __attribute__((ext_vector_type(16))) __bf16 v16bf;
typedef __attribute__((ext_vector_type(8)))  __bf16 v8bf;
typedef __attribute__((ext_vector_type(8)))  float  v8f;
typedef __attribute__((ext_vector_type(4)))  unsigned int u32x4;
typedef __attribute__((ext_vector_type(8)))  int i32x8;
typedef __attribute__((ext_vector_type(4)))  int i32x4;

#if defined(__AMDGCN__) && __has_builtin(__builtin_amdgcn_tensor_load_to_lds) && \
    __has_builtin(__builtin_amdgcn_s_wait_tensorcnt)
#define USE_TDM 1
#else
#define USE_TDM 0
#endif

#define T_DIM 4096
#define H_DIM 4096
#define F_DIM 14336

#define TILE_M 128
#define TILE_N 128
#define TILE_K 32   // one WMMA K-step

static __device__ inline v8f wmma_bf16(v16bf a, v16bf b, v8f c) {
  return __builtin_amdgcn_wmma_f32_16x16x32_bf16(false, a, false, b,
                                                 (short)0, c, false, false);
}

// 16x32 bf16 fragment from an LDS tile laid out [rows][32 K halves].
// Lane = row (mod 16); lane>>4 selects +8 K-half; VGPR0-3: K=khalf+0..7,
// VGPR4-7: K=16+khalf+0..7 (ISA 7.12.2). B mirrored with lane = N column.
static __device__ inline v16bf ld_frag(const __bf16* tile, int row, int khalf) {
  const v8bf* p = (const v8bf*)(tile + row * TILE_K + khalf);
  v8bf lo = p[0];
  v8bf hi = p[2];   // +16 halves
  return __builtin_shufflevector(lo, hi, 0, 1, 2, 3, 4, 5, 6, 7,
                                          8, 9, 10, 11, 12, 13, 14, 15);
}

static __device__ inline unsigned pack_bf16(float a, float b) {
  union { __bf16 h[2]; unsigned u; } p;
  p.h[0] = (__bf16)a;   // low 16 bits  -> K
  p.h[1] = (__bf16)b;   // high 16 bits -> K+1
  return p.u;
}

#if USE_TDM
// Build a 2D tensor D# and issue TENSOR_LOAD_TO_LDS (6-arg builtin on this
// toolchain: g0, g1, g2, g3, g1-of-2nd-desc, cpol — trailing groups zeroed).
// tile: tile_k elements (row, data_size=2B) x tile_rows rows; row stride
// stride0 elements; LDS destination is tightly packed (tile_rows*tile_k*2 B).
static __device__ inline void tdm_load_2d(unsigned lds_off, const void* gaddr,
                                          unsigned tile_k, unsigned tile_rows,
                                          unsigned tensor_d0, unsigned tensor_d1,
                                          unsigned long long stride0) {
  const unsigned long long ga = (unsigned long long)(uintptr_t)gaddr;
  u32x4 g0;
  g0[0] = 1u;                                              // count=1, user D#
  g0[1] = lds_off;                                         // lds_addr (bytes)
  g0[2] = (unsigned)(ga & 0xFFFFFFFFu);                    // global_addr[31:0]
  g0[3] = (unsigned)((ga >> 32) & 0x1FFFFFFu) | (2u << 30); // [56:32] | type=2
  i32x8 g1;
  g1[0] = (int)(1u << 16);                                 // data_size=1 (2B)
  g1[1] = (int)((tensor_d0 & 0xFFFFu) << 16);              // td0[15:0]
  g1[2] = (int)((tensor_d0 >> 16) | ((tensor_d1 & 0xFFFFu) << 16));
  g1[3] = (int)((tensor_d1 >> 16) | (tile_k << 16));       // td1 hi | tile_dim0
  g1[4] = (int)(tile_rows & 0xFFFFu);                      // tile_dim1, dim2=0
  g1[5] = (int)(stride0 & 0xFFFFFFFFu);                    // stride0 lo
  g1[6] = (int)((stride0 >> 32) & 0xFFFFu);                // stride0 hi
  g1[7] = 0;
  const i32x4 z4 = {0, 0, 0, 0};                           // groups 2/3 unused
  const i32x8 z8 = {0, 0, 0, 0, 0, 0, 0, 0};
  __builtin_amdgcn_tensor_load_to_lds(g0, g1, z4, z4, z8, 0);
}

static __device__ inline unsigned lds_offset(const void* p) {
  // generic pointer to LDS = SHARED_BASE | offset; low 32 bits = byte offset
  return (unsigned)(uintptr_t)p;
}
#endif

// ---------------------------------------------------------------------------
// Pass 1: dual GEMM (gate/up) + SwiGLU epilogue.  Grid: (F/128, T/128).
// 8 waves = 2(M) x 4(N); each wave 64x32 = 4x2 fragments, A frags shared
// between the w1 and v1 GEMMs.  Double-buffered LDS.
// ---------------------------------------------------------------------------
__global__ __launch_bounds__(256) void swiglu_gateup_kernel(
    const float* __restrict__ x,   // [T, H]
    const float* __restrict__ w1,  // [H, F]
    const float* __restrict__ v1,  // [H, F]
    __bf16* __restrict__ hbuf)     // [T, F] bf16
{
  __shared__ __bf16 As[2][TILE_M * TILE_K];  // [m][k]
  __shared__ __bf16 Bw[2][TILE_N * TILE_K];  // [n][k] (transposed on store)
  __shared__ __bf16 Bv[2][TILE_N * TILE_K];  // [n][k]

  const int tid  = threadIdx.x;
  const int lane = tid & 31;
  const int wave = tid >> 5;
  const int wm = wave & 1;
  const int wn = wave >> 1;
  const int tm = blockIdx.y * TILE_M;
  const int tn = blockIdx.x * TILE_N;
  const int lrow  = lane & 15;
  const int khalf = (lane >> 4) * 8;

  v8f accG[4][2], accU[4][2];
  const v8f vzero = {0.f, 0.f, 0.f, 0.f, 0.f, 0.f, 0.f, 0.f};
#pragma unroll
  for (int i = 0; i < 4; ++i)
#pragma unroll
    for (int j = 0; j < 2; ++j) { accG[i][j] = vzero; accU[i][j] = vzero; }

  // A loader: row = tid>>1, 16 floats at col (tid&1)*16.
  const int ar = tid >> 1;
  const int ac = (tid & 1) * 16;
  // B loader: n fastest across lanes (coalesced), each thread owns a K-pair:
  const int bn = (tid & 15) * 8;       // 8 consecutive n columns
  const int bkk = (tid >> 4) * 2;      // k, k+1 rows

  float sa[16], sw0[8], sw1[8], sv0[8], sv1[8];

  auto stage = [&](int k0) {
    const float* gpa = x + (size_t)(tm + ar) * H_DIM + (k0 + ac);
    __builtin_prefetch(gpa + TILE_K, 0, 3);
#pragma unroll
    for (int q = 0; q < 4; ++q) {
      const float4 f = ((const float4*)gpa)[q];
      sa[4 * q + 0] = f.x; sa[4 * q + 1] = f.y; sa[4 * q + 2] = f.z; sa[4 * q + 3] = f.w;
    }
    const float* gw0 = w1 + (size_t)(k0 + bkk) * F_DIM + (tn + bn);
    const float* gv0 = v1 + (size_t)(k0 + bkk) * F_DIM + (tn + bn);
    __builtin_prefetch(gw0 + (size_t)TILE_K * F_DIM, 0, 3);
    __builtin_prefetch(gv0 + (size_t)TILE_K * F_DIM, 0, 3);
#pragma unroll
    for (int q = 0; q < 2; ++q) {
      float4 f;
      f = ((const float4*)gw0)[q];
      sw0[4 * q + 0] = f.x; sw0[4 * q + 1] = f.y; sw0[4 * q + 2] = f.z; sw0[4 * q + 3] = f.w;
      f = ((const float4*)(gw0 + F_DIM))[q];
      sw1[4 * q + 0] = f.x; sw1[4 * q + 1] = f.y; sw1[4 * q + 2] = f.z; sw1[4 * q + 3] = f.w;
      f = ((const float4*)gv0)[q];
      sv0[4 * q + 0] = f.x; sv0[4 * q + 1] = f.y; sv0[4 * q + 2] = f.z; sv0[4 * q + 3] = f.w;
      f = ((const float4*)(gv0 + F_DIM))[q];
      sv1[4 * q + 0] = f.x; sv1[4 * q + 1] = f.y; sv1[4 * q + 2] = f.z; sv1[4 * q + 3] = f.w;
    }
  };

  auto commit = [&](int p) {
    v8bf lo, hi;
#pragma unroll
    for (int i = 0; i < 8; ++i) { lo[i] = (__bf16)sa[i]; hi[i] = (__bf16)sa[8 + i]; }
    v8bf* sp = (v8bf*)(As[p] + ar * TILE_K + ac);
    sp[0] = lo;
    sp[1] = hi;
    unsigned* bwp = (unsigned*)Bw[p];
    unsigned* bvp = (unsigned*)Bv[p];
    const int kword = bkk >> 1;        // dword index in a 32-half row
#pragma unroll
    for (int j = 0; j < 8; ++j) {
      bwp[(bn + j) * (TILE_K / 2) + kword] = pack_bf16(sw0[j], sw1[j]);
      bvp[(bn + j) * (TILE_K / 2) + kword] = pack_bf16(sv0[j], sv1[j]);
    }
  };

  auto compute = [&](int p) {
    v16bf afr[4];
#pragma unroll
    for (int fm = 0; fm < 4; ++fm)
      afr[fm] = ld_frag(As[p], wm * 64 + fm * 16 + lrow, khalf);
#pragma unroll
    for (int fn = 0; fn < 2; ++fn) {
      const v16bf bw = ld_frag(Bw[p], wn * 32 + fn * 16 + lrow, khalf);
      const v16bf bv = ld_frag(Bv[p], wn * 32 + fn * 16 + lrow, khalf);
#pragma unroll
      for (int fm = 0; fm < 4; ++fm) {
        accG[fm][fn] = wmma_bf16(afr[fm], bw, accG[fm][fn]);
        accU[fm][fn] = wmma_bf16(afr[fm], bv, accU[fm][fn]);
      }
    }
  };

  // ---- ping-pong main loop: one barrier per K-step ----
  stage(0);
  commit(0);
  __syncthreads();
  int p = 0;
  for (int k0 = 0; k0 < H_DIM; k0 += TILE_K) {
    const bool more = (k0 + TILE_K) < H_DIM;
    if (more) stage(k0 + TILE_K);   // global loads overlap compute below
    compute(p);
    if (more) commit(p ^ 1);        // write the idle buffer
    __syncthreads();
    p ^= 1;
  }

  // ---- SwiGLU epilogue: silu(gate) * up -> bf16 workspace ----
#pragma unroll
  for (int fm = 0; fm < 4; ++fm) {
#pragma unroll
    for (int fn = 0; fn < 2; ++fn) {
      const int col = tn + wn * 32 + fn * 16 + lrow;
#pragma unroll
      for (int r = 0; r < 8; ++r) {
        const int row = tm + wm * 64 + fm * 16 + (lane >> 4) * 8 + r;
        const float g = accG[fm][fn][r];
        const float u = accU[fm][fn][r];
        const float s = g / (1.0f + __expf(-g));  // silu
        hbuf[(size_t)row * F_DIM + col] = (__bf16)(s * u);
      }
    }
  }
}

// ---------------------------------------------------------------------------
// Pass 2: out = h @ w2^T.  NT GEMM, everything K-contiguous.  A tiles are
// bf16 block copies -> Tensor Data Mover when available.  Grid: (H/128, T/128).
// ---------------------------------------------------------------------------
__global__ __launch_bounds__(256) void swiglu_down_kernel(
    const __bf16* __restrict__ hbuf,  // [T, F] bf16
    const float* __restrict__ w2,     // [H, F]
    float* __restrict__ out)          // [T, H]
{
  __shared__ __bf16 As[2][TILE_M * TILE_K];  // [m][k]
  __shared__ __bf16 Bs[2][TILE_N * TILE_K];  // [n][k]

  const int tid  = threadIdx.x;
  const int lane = tid & 31;
  const int wave = tid >> 5;
  const int wm = wave & 1;
  const int wn = wave >> 1;
  const int tm = blockIdx.y * TILE_M;
  const int tn = blockIdx.x * TILE_N;
  const int lrow  = lane & 15;
  const int khalf = (lane >> 4) * 8;

  v8f acc[4][2];
  const v8f vzero = {0.f, 0.f, 0.f, 0.f, 0.f, 0.f, 0.f, 0.f};
#pragma unroll
  for (int i = 0; i < 4; ++i)
#pragma unroll
    for (int j = 0; j < 2; ++j) acc[i][j] = vzero;

  const int br = tid >> 1;          // row 0..127
  const int bc = (tid & 1) * 16;    // k offset 0/16

  float sb[16];
#if !USE_TDM
  uint4 ua0, ua1;
#endif

  auto stageB = [&](int k0) {
    const float* gp = w2 + (size_t)(tn + br) * F_DIM + (k0 + bc);
    __builtin_prefetch(gp + TILE_K, 0, 3);
#pragma unroll
    for (int q = 0; q < 4; ++q) {
      const float4 f = ((const float4*)gp)[q];
      sb[4 * q + 0] = f.x; sb[4 * q + 1] = f.y; sb[4 * q + 2] = f.z; sb[4 * q + 3] = f.w;
    }
  };
  auto commitB = [&](int p) {
    v8bf lo, hi;
#pragma unroll
    for (int i = 0; i < 8; ++i) { lo[i] = (__bf16)sb[i]; hi[i] = (__bf16)sb[8 + i]; }
    v8bf* sp = (v8bf*)(Bs[p] + br * TILE_K + bc);
    sp[0] = lo;
    sp[1] = hi;
  };

#if USE_TDM
  auto loadA = [&](int p, int k0) {
    if (wave == 0)
      tdm_load_2d(lds_offset(As[p]),
                  (const void*)(hbuf + (size_t)tm * F_DIM + k0),
                  TILE_K, TILE_M, F_DIM, T_DIM, (unsigned long long)F_DIM);
  };
#else
  auto stageA = [&](int k0) {
    const uint4* gp = (const uint4*)(hbuf + (size_t)(tm + br) * F_DIM + (k0 + bc));
    __builtin_prefetch(gp + 2, 0, 3);
    ua0 = gp[0];
    ua1 = gp[1];
  };
  auto commitA = [&](int p) {
    uint4* sp = (uint4*)(As[p] + br * TILE_K + bc);
    sp[0] = ua0;
    sp[1] = ua1;
  };
#endif

  auto compute = [&](int p) {
    v16bf afr[4];
#pragma unroll
    for (int fm = 0; fm < 4; ++fm)
      afr[fm] = ld_frag(As[p], wm * 64 + fm * 16 + lrow, khalf);
#pragma unroll
    for (int fn = 0; fn < 2; ++fn) {
      const v16bf bfr = ld_frag(Bs[p], wn * 32 + fn * 16 + lrow, khalf);
#pragma unroll
      for (int fm = 0; fm < 4; ++fm)
        acc[fm][fn] = wmma_bf16(afr[fm], bfr, acc[fm][fn]);
    }
  };

  // ---- prologue ----
#if USE_TDM
  loadA(0, 0);
#else
  stageA(0);
  commitA(0);
#endif
  stageB(0);
  commitB(0);
#if USE_TDM
  __builtin_amdgcn_s_wait_tensorcnt(0);
#endif
  __syncthreads();

  int p = 0;
  for (int k0 = 0; k0 < F_DIM; k0 += TILE_K) {
    const bool more = (k0 + TILE_K) < F_DIM;
    if (more) {
#if USE_TDM
      loadA(p ^ 1, k0 + TILE_K);    // TDM fills idle buffer during compute
#else
      stageA(k0 + TILE_K);
#endif
      stageB(k0 + TILE_K);
    }
    compute(p);
    if (more) {
      commitB(p ^ 1);
#if !USE_TDM
      commitA(p ^ 1);
#endif
    }
#if USE_TDM
    __builtin_amdgcn_s_wait_tensorcnt(0);  // no-op for waves with TENSORcnt==0
#endif
    __syncthreads();
    p ^= 1;
  }

#pragma unroll
  for (int fm = 0; fm < 4; ++fm) {
#pragma unroll
    for (int fn = 0; fn < 2; ++fn) {
      const int col = tn + wn * 32 + fn * 16 + lrow;
#pragma unroll
      for (int r = 0; r < 8; ++r) {
        const int row = tm + wm * 64 + fm * 16 + (lane >> 4) * 8 + r;
        out[(size_t)row * H_DIM + col] = acc[fm][fn][r];
      }
    }
  }
}

// ---------------------------------------------------------------------------
extern "C" void kernel_launch(void* const* d_in, const int* in_sizes, int n_in,
                              void* d_out, int out_size, void* d_ws, size_t ws_size,
                              hipStream_t stream) {
  const float* x  = (const float*)d_in[0];  // [T, H]
  const float* w1 = (const float*)d_in[1];  // [H, F]
  const float* v1 = (const float*)d_in[2];  // [H, F]
  const float* w2 = (const float*)d_in[3];  // [H, F]
  float* out = (float*)d_out;               // [T, H]
  __bf16* hbuf = (__bf16*)d_ws;             // [T, F] bf16 intermediate (112 MB)

  dim3 g1(F_DIM / TILE_N, T_DIM / TILE_M);  // (112, 32)
  swiglu_gateup_kernel<<<g1, 256, 0, stream>>>(x, w1, v1, hbuf);

  dim3 g2(H_DIM / TILE_N, T_DIM / TILE_M);  // (32, 32)
  swiglu_down_kernel<<<g2, 256, 0, stream>>>(hbuf, w2, out);
}
